// BitLinear_22162031248001
// MI455X (gfx1250) — matrix operand order repaired and checked by reading, not compiled, not run on hardware
//
#include <hip/hip_runtime.h>

typedef __attribute__((ext_vector_type(16))) _Float16 v16h;
typedef __attribute__((ext_vector_type(8)))  float    v8f;

#define TOKENS 64
#define INF    4096
#define OUTF   11008
#define BLK_N  128      // 8 waves * 16 columns
#define KTILE  64       // K staged per barrier (2 WMMA K-steps)

// ---------------------------------------------------------------------------
// Pass 1: x f32 -> f16 into workspace (512 KB)
// ---------------------------------------------------------------------------
__global__ void bitlinear_cvt_f16(const float* __restrict__ x,
                                  _Float16* __restrict__ xh, int n) {
  int i = blockIdx.x * blockDim.x + threadIdx.x;
  if (i < n) xh[i] = (_Float16)x[i];
}

// ---------------------------------------------------------------------------
// Pass 2: WMMA GEMM with on-the-fly 1-bit -> f16 (+/-1) unpack.
// Block: 256 threads = 8 waves. Block tile: N=128 (16 per wave), M=64 (4
// accumulators per wave), full K=4096. Each bp byte is read exactly once.
// Per K-tile of 64: one LDS stage + barrier, then 2 x (8 A-frag ds_loads ->
// B unpack VALU hides LDS latency -> 4 back-to-back WMMAs).
// ---------------------------------------------------------------------------
__launch_bounds__(256)
__global__ void bitlinear_wmma(const _Float16* __restrict__ xh,
                               const int* __restrict__ bp,
                               const float* __restrict__ scale,
                               float* __restrict__ out) {
  __shared__ __align__(16) _Float16 As[TOKENS * KTILE];   // 64x64 f16 = 8 KB

  const int tid  = threadIdx.x;
  const int lane = tid & 31;
  const int wave = tid >> 5;
  const bool hi  = lane >= 16;

  const int ncol = lane & 15;
  const int o    = blockIdx.x * BLK_N + wave * 16 + ncol;  // weight row / out col
  const float s  = scale[o];

  // bp row: 4 ints per 32-bit K-step; low half-wave ints 0..1 (K0-15),
  // high half-wave ints 2..3 (K16-31).
  const int* bprow = bp + (size_t)o * (INF / 8) + (hi ? 2 : 0);

  // LDS staging: thread copies 32 B of the 64x64 x-tile per K-tile.
  const int cm = tid >> 2;           // row 0..63
  const int cs = (tid & 3) * 16;     // half-offset 0,16,32,48

  // A-fragment addressing (contiguous K order, same as B): lanes<16 read K {0..15},
  // lanes>=16 read K {16..31} of their row (relative to each K-step).
  const int am = lane & 15;
  const int ak = hi ? 16 : 0;

  v8f acc[4] = {v8f{}, v8f{}, v8f{}, v8f{}};

  for (int kb = 0; kb < INF; kb += KTILE) {
    // prefetch next K-tile's bp stream (64 B ahead) -> global_prefetch_b8
    __builtin_prefetch((const void*)(bprow + (kb >> 3) + 16), 0, 1);

    // stage 64x64 x tile into LDS (two 16B chunks per thread)
    {
      const _Float16* gsrc = xh + (size_t)cm * INF + kb + cs;
      _Float16*       ldst = &As[cm * KTILE + cs];
      *(uint4*)(ldst)     = *(const uint4*)(gsrc);
      *(uint4*)(ldst + 8) = *(const uint4*)(gsrc + 8);
    }
    __syncthreads();

#pragma unroll
    for (int ks = 0; ks < KTILE; ks += 32) {
      // ---- preload all 4 A fragments (8 ds_load_b128) -------------------
      union AF { v16h v; uint4 q[2]; } A[4];
#pragma unroll
      for (int mt = 0; mt < 4; ++mt) {
        const int row = mt * 16 + am;
        A[mt].q[0] = *(const uint4*)&As[row * KTILE + ks + ak];       // K 0..7 / 16..23
        A[mt].q[1] = *(const uint4*)&As[row * KTILE + ks + ak + 8];   // K 8..15 / 24..31
      }

      // ---- B fragment: 16 sign bits -> 8 packed f16 pairs of +/-1 -------
      // value = bit ? +1 : -1; f16(+1)=0x3C00, sign set when bit==0, so work
      // on inverted bits. Reference is MSB-first within each byte. This VALU
      // block overlaps the ds_load latency above.
      int2 w = *(const int2*)(bprow + ((kb + ks) >> 3));
      unsigned w0 = ~(unsigned)w.x;   // K rel 0..7   (or 16..23 for hi lanes)
      unsigned w1 = ~(unsigned)w.y;   // K rel 8..15  (or 24..31)

      union { v16h v; unsigned u[8]; } B;
#pragma unroll
      for (int j = 0; j < 8; ++j) {
        unsigned b  = (j < 4) ? w0 : w1;
        int      jp = (j & 3) * 2;                       // 2*j' within byte
        unsigned p  = 0x3C003C00u;                       // +1,+1
        p |= (b << (8  + jp)) & 0x00008000u;             // K=2j'   -> low sign
        p |= (b << (25 + jp)) & 0x80000000u;             // K=2j'+1 -> high sign
        B.u[j] = p;
      }

      // ---- 4 back-to-back WMMAs sharing this B fragment -----------------
#pragma unroll
      for (int mt = 0; mt < 4; ++mt) {
        acc[mt] = __builtin_amdgcn_wmma_f32_16x16x32_f16(
            /*neg_a=*/false, A[mt].v, /*neg_b=*/false, B.v,
            /*c_mod=*/(short)0, acc[mt], /*reuse_a=*/false, /*reuse_b=*/false);
      }
    }
    __syncthreads();
  }

  // Epilogue: D layout -> lanes<16 carry M=r, lanes>=16 carry M=8+r.
#pragma unroll
  for (int mt = 0; mt < 4; ++mt) {
#pragma unroll
    for (int r = 0; r < 8; ++r) {
      const int m = mt * 16 + r + (hi ? 8 : 0);
      out[(size_t)m * OUTF + o] = acc[mt][r] * s;
    }
  }
}

// ---------------------------------------------------------------------------
extern "C" void kernel_launch(void* const* d_in, const int* in_sizes, int n_in,
                              void* d_out, int out_size, void* d_ws, size_t ws_size,
                              hipStream_t stream) {
  const float* x     = (const float*)d_in[0];
  const int*   bp    = (const int*)d_in[1];
  const float* scale = (const float*)d_in[2];
  float*       out   = (float*)d_out;
  _Float16*    xh    = (_Float16*)d_ws;   // 64*4096*2 = 512 KB

  const int nx = TOKENS * INF;
  bitlinear_cvt_f16<<<(nx + 255) / 256, 256, 0, stream>>>(x, xh, nx);
  bitlinear_wmma<<<OUTF / BLK_N, 256, 0, stream>>>(xh, bp, scale, out);
}
